// _HFRefMoE_19000935317689
// MI455X (gfx1250) — compile-verified, matrix operand order checked by reading, not run on hardware
//
#include <hip/hip_runtime.h>

// Problem constants (B=1, S=2048)
#define TOK   2048   // tokens
#define HID   1024   // hidden
#define IEX   512    // routed expert intermediate
#define NEXP  8      // routed experts
#define NSHI  1024   // shared intermediate (NSH * I)

typedef __attribute__((ext_vector_type(16))) __bf16 v16bf;
typedef __attribute__((ext_vector_type(8)))  __bf16 v8bf;
typedef __attribute__((ext_vector_type(4)))  __bf16 v4bf;
typedef __attribute__((ext_vector_type(8)))  float  v8f;
typedef __attribute__((ext_vector_type(4)))  float  v4f;

// Fast sigmoid: v_exp_f32 + v_rcp_f32 (no IEEE divide sequence).
__device__ __forceinline__ float fast_sigmoid(float v) {
  return __builtin_amdgcn_rcpf(1.0f + __expf(-v));
}

// Build a 16x32 bf16 A/B fragment for one lane from a row stored [outer][K]
// in bf16.  ISA layout (CDNA5 ISA): lanes 0-15 hold K = kb+{0..7} and
// kb+{16..23}; lanes 16-31 hold K = kb+8+{0..7} and kb+8+{16..23}.  Caller
// folds the per-lane +8 into kbase.  Two 16-byte loads, no conversion.
__device__ __forceinline__ v16bf frag_bf16(const unsigned short* row, int kbase) {
  v8bf a = *(const v8bf*)(row + kbase);
  v8bf b = *(const v8bf*)(row + kbase + 16);
  v16bf r;
#pragma unroll
  for (int i = 0; i < 8; ++i) { r[i] = a[i]; r[i + 8] = b[i]; }
  return r;
}

// ---------------------------------------------------------------------------
// Kernel 0: bulk fp32 -> bf16 conversion (one pass; everything fits in L2,
// so this is a cheap bandwidth-bound prologue that strips all conversion
// VALU out of the GEMM inner loops).
// ---------------------------------------------------------------------------
__global__ __launch_bounds__(256)
void cvt_f32_to_bf16(const float* __restrict__ src,
                     unsigned short* __restrict__ dst, int n4) {
  int i = blockIdx.x * 256 + threadIdx.x;
  if (i >= n4) return;
  v4f v = *(const v4f*)(src + (size_t)i * 4);
  v4bf r;
  r[0] = (__bf16)v[0];
  r[1] = (__bf16)v[1];
  r[2] = (__bf16)v[2];
  r[3] = (__bf16)v[3];
  *(v4bf*)((__bf16*)dst + (size_t)i * 4) = r;
}

// ---------------------------------------------------------------------------
// Kernel 1: router.  scores = sigmoid(x @ rw^T); grouped top-2-of-4-groups,
// then top-2 experts; weights gathered from ORIGINAL scores, normalized.
// Writes dense combine weights cw[T, 8] (zero for unselected experts).
// ---------------------------------------------------------------------------
__global__ __launch_bounds__(256)
void moe_router(const float* __restrict__ x, const float* __restrict__ rw,
                const float* __restrict__ bias, float* __restrict__ cw) {
  int t = blockIdx.x * 256 + threadIdx.x;
  if (t >= TOK) return;
  float acc[NEXP];
#pragma unroll
  for (int e = 0; e < NEXP; ++e) acc[e] = 0.0f;
  const float* xr = x + (size_t)t * HID;
#pragma unroll 2
  for (int k = 0; k < HID; k += 4) {
    v4f xv = *(const v4f*)(xr + k);
#pragma unroll
    for (int e = 0; e < NEXP; ++e) {
      v4f wv = *(const v4f*)(rw + (size_t)e * HID + k);
      acc[e] += xv[0]*wv[0] + xv[1]*wv[1] + xv[2]*wv[2] + xv[3]*wv[3];
    }
  }
  float s[NEXP], sc[NEXP];
#pragma unroll
  for (int e = 0; e < NEXP; ++e) {
    s[e]  = fast_sigmoid(acc[e]);
    sc[e] = s[e] + bias[e];
  }
  float gs[4];
#pragma unroll
  for (int g = 0; g < 4; ++g) gs[g] = sc[2*g] + sc[2*g + 1];
  int g0 = 0; float b0 = gs[0];
#pragma unroll
  for (int g = 1; g < 4; ++g) if (gs[g] > b0) { b0 = gs[g]; g0 = g; }
  int g1 = -1; float b1 = -1e30f;
#pragma unroll
  for (int g = 0; g < 4; ++g) if (g != g0 && gs[g] > b1) { b1 = gs[g]; g1 = g; }
  float masked[NEXP];
#pragma unroll
  for (int e = 0; e < NEXP; ++e) {
    int grp = e >> 1;
    masked[e] = (grp == g0 || grp == g1) ? sc[e] : 0.0f;
  }
  int i0 = 0; float m0 = masked[0];
#pragma unroll
  for (int e = 1; e < NEXP; ++e) if (masked[e] > m0) { m0 = masked[e]; i0 = e; }
  int i1 = -1; float m1 = -1e30f;
#pragma unroll
  for (int e = 0; e < NEXP; ++e) if (e != i0 && masked[e] > m1) { m1 = masked[e]; i1 = e; }
  float w0 = s[i0], w1 = s[i1];
  float inv = __builtin_amdgcn_rcpf(w0 + w1 + 1e-20f);  // routed_scaling_factor = 1.0
  float o[NEXP];
#pragma unroll
  for (int e = 0; e < NEXP; ++e) o[e] = 0.0f;
  o[i0] += w0 * inv;
  o[i1] += w1 * inv;
#pragma unroll
  for (int e = 0; e < NEXP; ++e) cw[(size_t)t * NEXP + e] = o[e];
}

// ---------------------------------------------------------------------------
// Kernel 2: stage 1 — per-"expert" SwiGLU up path, all-bf16 operands.
// blockIdx.y in [0,10): 0..7 routed experts, 8..9 = halves of the shared
// expert.  x tile (16 x 1024 bf16) is DMA'd into LDS with async
// global->LDS b128 copies (ASYNCcnt), then 8 waves x 4 n-tiles run WMMA.
// ---------------------------------------------------------------------------
__global__ __launch_bounds__(256)
void moe_stage1(const unsigned short* __restrict__ xb,
                const unsigned short* __restrict__ wgb,
                const unsigned short* __restrict__ wub,
                const unsigned short* __restrict__ sgb,
                const unsigned short* __restrict__ sub,
                unsigned short* __restrict__ actR,    // [8][T][512] bf16
                unsigned short* __restrict__ actS) {  // [T][1024]   bf16
  __shared__ unsigned short xs[16][1032];             // 16 x 1024 bf16 (+16B pad)
  const int t0  = blockIdx.x * 16;
  const int e   = blockIdx.y;                         // 0..9
  const int tid = threadIdx.x;

  // Async DMA of the bf16 x tile into LDS: 2048 x 16-byte chunks.
  {
    unsigned int ldsbase = (unsigned int)(size_t)(&xs[0][0]);
    unsigned long long gbase =
        (unsigned long long)(size_t)xb + (unsigned long long)t0 * (HID * 2);
#pragma unroll
    for (int i = 0; i < 8; ++i) {
      int c   = tid + i * 256;          // chunk id, 128 chunks per row
      int row = c >> 7;
      int col = c & 127;
      unsigned int ldsoff = ldsbase + row * (1032 * 2) + col * 16;
      unsigned int voff   = row * (HID * 2) + col * 16;
      asm volatile("global_load_async_to_lds_b128 %0, %1, %2"
                   :: "v"(ldsoff), "v"(voff), "s"(gbase) : "memory");
    }
    asm volatile("s_wait_asynccnt 0" ::: "memory");
  }
  __syncthreads();

  const int wave  = tid >> 5;
  const int lane  = tid & 31;
  const int lrow  = lane & 15;
  const int koff  = (lane >> 4) << 3;                 // 0 or 8
  const int mbase = (lane >> 4) << 3;

  const unsigned short* Wg;
  const unsigned short* Wu;
  if (e < NEXP) { Wg = wgb + (size_t)e * IEX * HID;          Wu = wub + (size_t)e * IEX * HID; }
  else          { Wg = sgb + (size_t)(e - NEXP) * IEX * HID; Wu = sub + (size_t)(e - NEXP) * IEX * HID; }

  v8f g[4], u[4];
#pragma unroll
  for (int j = 0; j < 4; ++j) { g[j] = (v8f){}; u[j] = (v8f){}; }

  const int nbase = wave * 64;                        // 4 n-tiles of 16 per wave
  for (int k0 = 0; k0 < HID; k0 += 32) {
    const int kk = k0 + koff;
    v16bf a = frag_bf16(&xs[lrow][0], kk);
    v16bf bg[4], bu[4];
#pragma unroll
    for (int j = 0; j < 4; ++j) {
      int n = nbase + j * 16 + lrow;
      bg[j] = frag_bf16(Wg + (size_t)n * HID, kk);
      bu[j] = frag_bf16(Wu + (size_t)n * HID, kk);
    }
#pragma unroll
    for (int j = 0; j < 4; ++j) {
      g[j] = __builtin_amdgcn_wmma_f32_16x16x32_bf16(false, a, false, bg[j],
                                                     (short)0, g[j], false, false);
      u[j] = __builtin_amdgcn_wmma_f32_16x16x32_bf16(false, a, false, bu[j],
                                                     (short)0, u[j], false, false);
    }
  }

  // act = silu(g) * u ; store bf16.  D layout: VGPR i = row M (= i or i+8),
  // lane&15 = col N.  silu via v_exp + v_rcp (no IEEE div sequence).
#pragma unroll
  for (int j = 0; j < 4; ++j) {
    int n = nbase + j * 16 + lrow;
#pragma unroll
    for (int i = 0; i < 8; ++i) {
      float gv = g[j][i];
      float av = gv * fast_sigmoid(gv) * u[j][i];
      int t = t0 + mbase + i;
      unsigned short bits = __builtin_bit_cast(unsigned short, (__bf16)av);
      if (e < NEXP) actR[((size_t)e * TOK + t) * IEX + n] = bits;
      else          actS[(size_t)t * NSHI + (size_t)(e - NEXP) * IEX + n] = bits;
    }
  }
}

// ---------------------------------------------------------------------------
// Kernel 3: stage 2 — down projections + weighted combine, all-bf16 inputs.
// Each block: 16 tokens x 128 hidden cols; 8 waves, one 16x16 D-tile each.
// K-loop steps by 64 with both fragment pairs loaded up front for deeper
// load pipelining behind the chained WMMAs.
// out[t,h] = sum_e cw[t,e] * (act_e[t,:] @ wd_e[h,:]) + act_sh[t,:] @ sd[h,:]
// ---------------------------------------------------------------------------
__global__ __launch_bounds__(256)
void moe_stage2(const unsigned short* __restrict__ actR,
                const unsigned short* __restrict__ actS,
                const unsigned short* __restrict__ wdb,
                const unsigned short* __restrict__ sdb,
                const float* __restrict__ cw, float* __restrict__ out) {
  __shared__ float cws[16][NEXP];
  const int t0  = blockIdx.x * 16;
  const int tid = threadIdx.x;
  const int n0  = blockIdx.y * 128 + (tid >> 5) * 16;
  if (tid < 128) cws[tid >> 3][tid & 7] = cw[(size_t)(t0 + (tid >> 3)) * NEXP + (tid & 7)];
  __syncthreads();

  const int lane  = tid & 31;
  const int lrow  = lane & 15;
  const int koff  = (lane >> 4) << 3;
  const int mbase = (lane >> 4) << 3;

  v8f acc = (v8f){};
  for (int e = 0; e < NEXP; ++e) {
    v8f d = (v8f){};
    const unsigned short* arow = actR + ((size_t)e * TOK + t0 + lrow) * IEX;
    const unsigned short* brow = wdb  + ((size_t)e * HID + n0 + lrow) * IEX;
#pragma unroll 2
    for (int k0 = 0; k0 < IEX; k0 += 64) {
      v16bf a0 = frag_bf16(arow, k0 + koff);
      v16bf b0 = frag_bf16(brow, k0 + koff);
      v16bf a1 = frag_bf16(arow, k0 + 32 + koff);
      v16bf b1 = frag_bf16(brow, k0 + 32 + koff);
      d = __builtin_amdgcn_wmma_f32_16x16x32_bf16(false, a0, false, b0,
                                                  (short)0, d, false, false);
      d = __builtin_amdgcn_wmma_f32_16x16x32_bf16(false, a1, false, b1,
                                                  (short)0, d, false, false);
    }
#pragma unroll
    for (int i = 0; i < 8; ++i) acc[i] += cws[mbase + i][e] * d[i];
  }
  {
    v8f d = (v8f){};
    const unsigned short* arow = actS + (size_t)(t0 + lrow) * NSHI;
    const unsigned short* brow = sdb + (size_t)(n0 + lrow) * NSHI;
#pragma unroll 2
    for (int k0 = 0; k0 < NSHI; k0 += 64) {
      v16bf a0 = frag_bf16(arow, k0 + koff);
      v16bf b0 = frag_bf16(brow, k0 + koff);
      v16bf a1 = frag_bf16(arow, k0 + 32 + koff);
      v16bf b1 = frag_bf16(brow, k0 + 32 + koff);
      d = __builtin_amdgcn_wmma_f32_16x16x32_bf16(false, a0, false, b0,
                                                  (short)0, d, false, false);
      d = __builtin_amdgcn_wmma_f32_16x16x32_bf16(false, a1, false, b1,
                                                  (short)0, d, false, false);
    }
#pragma unroll
    for (int i = 0; i < 8; ++i) acc[i] += d[i];
  }
#pragma unroll
  for (int i = 0; i < 8; ++i)
    out[(size_t)(t0 + mbase + i) * HID + n0 + lrow] = acc[i];
}

// ---------------------------------------------------------------------------
extern "C" void kernel_launch(void* const* d_in, const int* in_sizes, int n_in,
                              void* d_out, int out_size, void* d_ws, size_t ws_size,
                              hipStream_t stream) {
  const float* x    = (const float*)d_in[0];   // [1,2048,1024]
  const float* rw   = (const float*)d_in[1];   // [8,1024]
  const float* bias = (const float*)d_in[2];   // [8]
  const float* wg   = (const float*)d_in[3];   // [8,512,1024]
  const float* wu   = (const float*)d_in[4];   // [8,512,1024]
  const float* wd   = (const float*)d_in[5];   // [8,1024,512]
  const float* sg   = (const float*)d_in[6];   // [1024,1024]
  const float* su   = (const float*)d_in[7];   // [1024,1024]
  const float* sd   = (const float*)d_in[8];   // [1024,1024]
  float* out = (float*)d_out;

  // Workspace layout (bytes):
  char* ws = (char*)d_ws;
  size_t off = 0;
  float*          cwp  = (float*)(ws + off);          off += (size_t)TOK * NEXP * 4;        // 64 KB
  unsigned short* xbb  = (unsigned short*)(ws + off); off += (size_t)TOK * HID * 2;         // 4 MB
  unsigned short* wgb  = (unsigned short*)(ws + off); off += (size_t)NEXP * IEX * HID * 2;  // 8 MB
  unsigned short* wub  = (unsigned short*)(ws + off); off += (size_t)NEXP * IEX * HID * 2;  // 8 MB
  unsigned short* wdb  = (unsigned short*)(ws + off); off += (size_t)NEXP * HID * IEX * 2;  // 8 MB
  unsigned short* sgb  = (unsigned short*)(ws + off); off += (size_t)NSHI * HID * 2;        // 2 MB
  unsigned short* sub  = (unsigned short*)(ws + off); off += (size_t)NSHI * HID * 2;        // 2 MB
  unsigned short* sdb  = (unsigned short*)(ws + off); off += (size_t)HID * NSHI * 2;        // 2 MB
  unsigned short* actR = (unsigned short*)(ws + off); off += (size_t)NEXP * TOK * IEX * 2;  // 16 MB
  unsigned short* actS = (unsigned short*)(ws + off); off += (size_t)TOK * NSHI * 2;        // 4 MB

  // fp32 -> bf16 conversion prologue (bandwidth-bound, fits in L2)
  auto cvt = [&](const float* s, unsigned short* d, size_t n) {
    int n4 = (int)(n / 4);
    cvt_f32_to_bf16<<<dim3((n4 + 255) / 256), dim3(256), 0, stream>>>(s, d, n4);
  };
  cvt(x,  xbb, (size_t)TOK * HID);
  cvt(wg, wgb, (size_t)NEXP * IEX * HID);
  cvt(wu, wub, (size_t)NEXP * IEX * HID);
  cvt(wd, wdb, (size_t)NEXP * HID * IEX);
  cvt(sg, sgb, (size_t)NSHI * HID);
  cvt(su, sub, (size_t)NSHI * HID);
  cvt(sd, sdb, (size_t)HID * NSHI);

  moe_router<<<dim3(TOK / 256), dim3(256), 0, stream>>>(x, rw, bias, cwp);
  moe_stage1<<<dim3(TOK / 16, NEXP + 2), dim3(256), 0, stream>>>(xbb, wgb, wub, sgb, sub, actR, actS);
  moe_stage2<<<dim3(TOK / 16, HID / 128), dim3(256), 0, stream>>>(actR, actS, wdb, sdb, cwp, out);
}
